// GPT_74586402062368
// MI455X (gfx1250) — matrix operand order repaired and checked by reading, not compiled, not run on hardware
//
#include <hip/hip_runtime.h>

// ---------------------------------------------------------------------------
// GPT-2 small forward for MI455X (gfx1250, wave32, WMMA).
// All GEMMs: v_wmma_f32_16x16x32_bf16, fp32 accumulate. Weights converted
// fp32 -> bf16 once per launch, pre-transposed to [N,K] so every GEMM uses a
// branch-free vectorized LDS staging path. Block tile 256x128 (8 waves),
// wave tile 64x64 = 16 WMMA per K-step. Software-pipelined: next K-tile is
// fetched into registers under the WMMA sequence; LDS is double-buffered so
// there is a single barrier per K-step.
// ---------------------------------------------------------------------------

typedef __attribute__((ext_vector_type(16))) __bf16 v16bf;
typedef __attribute__((ext_vector_type(8)))  float  v8f;

union Frag { v16bf v; uint4 q[2]; };

__device__ __forceinline__ unsigned short f2bf(float f) {
  unsigned int u = __builtin_bit_cast(unsigned int, f);
  unsigned int r = u + 0x7fffu + ((u >> 16) & 1u);   // round-to-nearest-even
  return (unsigned short)(r >> 16);
}
__device__ __forceinline__ unsigned int pack2(float a, float b) {
  return (unsigned int)f2bf(a) | ((unsigned int)f2bf(b) << 16);
}

enum {
  F_BIAS = 1, F_GELU = 2, F_RES = 4, F_BF16 = 8,
  F_CAUSAL = 16, F_BTRANS = 32, F_NGUARD = 64
};

// ---------------------------------------------------------------------------
// Batched WMMA GEMM: C[M,N] = A_bf16[M,K] @ B[K,N] (+epilogue).
// 256 threads (8 waves, 4x2). Block tile 256(M) x 128(N), K-step 32.
// M,K multiples of 256/32; N guarded only under F_NGUARD.
// Batch z = zb*batchH + zh with per-operand strides.
// ---------------------------------------------------------------------------
template <typename TB, int F>
__global__ __launch_bounds__(256, 1) void gemm_wmma(
    const unsigned short* __restrict__ Ag, const TB* __restrict__ Bg,
    const float* __restrict__ bias, float* __restrict__ Cf,
    unsigned short* __restrict__ Cb, const float* __restrict__ Rg,
    int N, int K, int lda, int ldb, int ldc,
    long long sAb, long long sAh, long long sBb, long long sBh,
    long long sCb, long long sCh, int batchH, float scale) {
  __shared__ unsigned short As[2][256][40];  // 80B rows: 16B-aligned chunks
  __shared__ unsigned short Bs[2][128][40];  // [N][K] (K contiguous)

  const int tid = threadIdx.x;
  const int z = blockIdx.z;
  const int zb = z / batchH, zh = z % batchH;
  Ag += zb * sAb + zh * sAh;
  Bg += zb * sBb + zh * sBh;
  const long long coff = (long long)zb * sCb + (long long)zh * sCh;

  const int m0 = blockIdx.y * 256;
  const int n0 = blockIdx.x * 128;
  const int w = tid >> 5, lane = tid & 31;
  const int wm = w >> 1, wn = w & 1;        // 4x2 wave grid
  const int half = lane >> 4, l15 = lane & 15;

  // staging thread mapping
  const int nn = tid >> 1, kk = (tid & 1) << 4;      // trans-B: [N][K] copies
  const int gn = n0 + nn;
  const int c4 = (tid & 31) << 2, kr = tid >> 5;     // non-trans fp32 B

  v8f acc[4][4];
#pragma unroll
  for (int i = 0; i < 4; ++i)
#pragma unroll
    for (int j = 0; j < 4; ++j) acc[i][j] = 0.0f;

  // ---- register staging for the software pipeline ------------------------
  uint4 aR0, aR1, aR2, aR3;                // A tile row (64B)
  uint4 bR0 = {0, 0, 0, 0}, bR1 = {0, 0, 0, 0};            // bf16 trans B
  float4 fR0 = {0, 0, 0, 0}, fR1 = {0, 0, 0, 0},
         fR2 = {0, 0, 0, 0}, fR3 = {0, 0, 0, 0};           // fp32 B

  auto fetch = [&](int kf) {
    const unsigned short* asrc = Ag + (size_t)(m0 + tid) * lda + kf;
    aR0 = *(const uint4*)asrc;
    aR1 = *(const uint4*)(asrc + 8);
    aR2 = *(const uint4*)(asrc + 16);
    aR3 = *(const uint4*)(asrc + 24);
    if (kf + 32 < K) __builtin_prefetch(asrc + 32, 0, 1);  // global_prefetch_b8
    if constexpr ((F & F_BTRANS) && sizeof(TB) == 2) {
      if (!(F & F_NGUARD) || gn < N) {
        const unsigned short* s = (const unsigned short*)Bg + (size_t)gn * ldb + kf + kk;
        bR0 = *(const uint4*)s;
        bR1 = *(const uint4*)(s + 8);
      } else {
        bR0 = {0, 0, 0, 0}; bR1 = {0, 0, 0, 0};
      }
    } else if constexpr ((F & F_BTRANS) && sizeof(TB) == 4) {
      if (!(F & F_NGUARD) || gn < N) {
        const float* s = (const float*)Bg + (size_t)gn * ldb + kf + kk;
        fR0 = *(const float4*)(s + 0);
        fR1 = *(const float4*)(s + 4);
        fR2 = *(const float4*)(s + 8);
        fR3 = *(const float4*)(s + 12);
      } else {
        fR0 = fR1 = fR2 = fR3 = {0, 0, 0, 0};
      }
    } else {
      const float* s = (const float*)Bg + (size_t)(kf + kr) * ldb + n0 + c4;
      fR0 = *(const float4*)s;
      fR1 = *(const float4*)(s + (size_t)8 * ldb);
      fR2 = *(const float4*)(s + (size_t)16 * ldb);
      fR3 = *(const float4*)(s + (size_t)24 * ldb);
    }
  };

  fetch(0);

  int p = 0;
  for (int k0 = 0; k0 < K; k0 += 32, p ^= 1) {
    // ---- store staged registers into LDS buffer p -----------------------
    *(uint4*)&As[p][tid][0]  = aR0;
    *(uint4*)&As[p][tid][8]  = aR1;
    *(uint4*)&As[p][tid][16] = aR2;
    *(uint4*)&As[p][tid][24] = aR3;
    if constexpr ((F & F_BTRANS) && sizeof(TB) == 2) {
      *(uint4*)&Bs[p][nn][kk]     = bR0;
      *(uint4*)&Bs[p][nn][kk + 8] = bR1;
    } else if constexpr ((F & F_BTRANS) && sizeof(TB) == 4) {
      uint4 q0, q1;
      q0.x = pack2(fR0.x, fR0.y); q0.y = pack2(fR0.z, fR0.w);
      q0.z = pack2(fR1.x, fR1.y); q0.w = pack2(fR1.z, fR1.w);
      q1.x = pack2(fR2.x, fR2.y); q1.y = pack2(fR2.z, fR2.w);
      q1.z = pack2(fR3.x, fR3.y); q1.w = pack2(fR3.z, fR3.w);
      *(uint4*)&Bs[p][nn][kk]     = q0;
      *(uint4*)&Bs[p][nn][kk + 8] = q1;
    } else {
      Bs[p][c4 + 0][kr]      = f2bf(fR0.x);
      Bs[p][c4 + 1][kr]      = f2bf(fR0.y);
      Bs[p][c4 + 2][kr]      = f2bf(fR0.z);
      Bs[p][c4 + 3][kr]      = f2bf(fR0.w);
      Bs[p][c4 + 0][kr + 8]  = f2bf(fR1.x);
      Bs[p][c4 + 1][kr + 8]  = f2bf(fR1.y);
      Bs[p][c4 + 2][kr + 8]  = f2bf(fR1.z);
      Bs[p][c4 + 3][kr + 8]  = f2bf(fR1.w);
      Bs[p][c4 + 0][kr + 16] = f2bf(fR2.x);
      Bs[p][c4 + 1][kr + 16] = f2bf(fR2.y);
      Bs[p][c4 + 2][kr + 16] = f2bf(fR2.z);
      Bs[p][c4 + 3][kr + 16] = f2bf(fR2.w);
      Bs[p][c4 + 0][kr + 24] = f2bf(fR3.x);
      Bs[p][c4 + 1][kr + 24] = f2bf(fR3.y);
      Bs[p][c4 + 2][kr + 24] = f2bf(fR3.z);
      Bs[p][c4 + 3][kr + 24] = f2bf(fR3.w);
    }
    // ---- launch next tile's global loads under this tile's compute ------
    if (k0 + 32 < K) fetch(k0 + 32);
    __syncthreads();   // LDS[p] visible; prior readers of LDS[p^1] done

    // ---- fragments: contiguous k = 16*half + s, same order for A and B --------------------------------------
    // A 16-bit 16x32: per lane K range (lane>=16)*16 .. +15
    // B 16-bit 32x16 (LDS [N][K]): per lane K range (lane>=16)*16 .. +15
    Frag bfr[4];
    const int ca = half * 16, kb2 = half * 16;
#pragma unroll
    for (int sn = 0; sn < 4; ++sn) {
      const int c = wn * 64 + sn * 16 + l15;
      bfr[sn].q[0] = *(const uint4*)&Bs[p][c][kb2];
      bfr[sn].q[1] = *(const uint4*)&Bs[p][c][kb2 + 8];
    }
#pragma unroll
    for (int sm = 0; sm < 4; ++sm) {
      Frag a;
      const int r = wm * 64 + sm * 16 + l15;
      a.q[0] = *(const uint4*)&As[p][r][ca];
      a.q[1] = *(const uint4*)&As[p][r][ca + 8];
#pragma unroll
      for (int sn = 0; sn < 4; ++sn)
        acc[sm][sn] = __builtin_amdgcn_wmma_f32_16x16x32_bf16(
            false, a.v, false, bfr[sn].v, (short)0, acc[sm][sn], false, false);
    }
  }

  // ---- epilogue: D layout -> row = base + half*8 + i, col = base + l15 ----
#pragma unroll
  for (int sm = 0; sm < 4; ++sm)
#pragma unroll
    for (int sn = 0; sn < 4; ++sn) {
      const int col = n0 + wn * 64 + sn * 16 + l15;
      if ((F & F_NGUARD) && col >= N) continue;
      float bv = 0.0f;
      if constexpr (F & F_BIAS) bv = bias[col];
#pragma unroll
      for (int i = 0; i < 8; ++i) {
        const int row = m0 + wm * 64 + sm * 16 + half * 8 + i;
        float v = acc[sm][sn][i] + bv;
        if constexpr (F & F_CAUSAL) v = (col <= row) ? v * scale : -1.0e30f;
        if constexpr (F & F_GELU)
          v = 0.5f * v * (1.0f + erff(v * 0.70710678118654752f));
        const long long idx = coff + (long long)row * ldc + col;
        if constexpr (F & F_RES) v += Rg[idx];
        if constexpr (F & F_BF16) Cb[idx] = f2bf(v);
        else                      Cf[idx] = v;
      }
    }
}

// ---------------------------------------------------------------------------
// One-time fp32 -> bf16 transposed weight conversion: W[K,N] -> wT[N,K].
// 32x32 tiles via LDS. Batched over layers (blockIdx.z).
// ---------------------------------------------------------------------------
__global__ __launch_bounds__(256) void transpose_cvt_f32(
    const float* __restrict__ W, unsigned short* __restrict__ wT,
    int K, int N, long long sIn, long long sOut) {
  __shared__ unsigned short t[32][33];
  W += (long long)blockIdx.z * sIn;
  wT += (long long)blockIdx.z * sOut;
  const int n0 = blockIdx.x * 32, k0 = blockIdx.y * 32;
  const int r = threadIdx.x >> 3;
  const int c4 = (threadIdx.x & 7) << 2;
  const float4 f = *(const float4*)&W[(size_t)(k0 + r) * N + n0 + c4];
  t[r][c4 + 0] = f2bf(f.x); t[r][c4 + 1] = f2bf(f.y);
  t[r][c4 + 2] = f2bf(f.z); t[r][c4 + 3] = f2bf(f.w);
  __syncthreads();
  union { uint2 u; unsigned short s[4]; } o;
  o.s[0] = t[c4 + 0][r]; o.s[1] = t[c4 + 1][r];
  o.s[2] = t[c4 + 2][r]; o.s[3] = t[c4 + 3][r];
  *(uint2*)&wT[(size_t)(n0 + r) * K + k0 + c4] = o.u;
}

// Straight fp32 -> bf16 copy (lm_head is already [N,K]).  n % 4 == 0.
__global__ __launch_bounds__(256) void copy_cvt_f32(
    const float* __restrict__ src, unsigned short* __restrict__ dst, int n4) {
  const int i = blockIdx.x * 256 + threadIdx.x;
  if (i >= n4) return;
  const float4 f = *(const float4*)&src[(size_t)i * 4];
  uint2 o; o.x = pack2(f.x, f.y); o.y = pack2(f.z, f.w);
  *(uint2*)&dst[(size_t)i * 4] = o;
}

// bf16 transpose of V: qkv[:, 2D + h*64 .. ][key][dh] -> vT[b,h][dh][key].
__global__ __launch_bounds__(256) void transpose_v(
    const unsigned short* __restrict__ qkv, unsigned short* __restrict__ vT) {
  __shared__ unsigned short t[32][33];
  const int z = blockIdx.z, b = z / 12, h = z % 12;
  const unsigned short* src = qkv + (size_t)b * 1024 * 2304 + 1536 + h * 64;
  unsigned short* dst = vT + (size_t)z * 64 * 1024;
  const int n0 = blockIdx.x * 32, k0 = blockIdx.y * 32;  // n = dh, k = key
  const int r = threadIdx.x >> 3;
  const int c4 = (threadIdx.x & 7) << 2;
  union { uint2 u; unsigned short s[4]; } io;
  io.u = *(const uint2*)&src[(size_t)(k0 + r) * 2304 + n0 + c4];
  t[r][c4 + 0] = io.s[0]; t[r][c4 + 1] = io.s[1];
  t[r][c4 + 2] = io.s[2]; t[r][c4 + 3] = io.s[3];
  __syncthreads();
  io.s[0] = t[c4 + 0][r]; io.s[1] = t[c4 + 1][r];
  io.s[2] = t[c4 + 2][r]; io.s[3] = t[c4 + 3][r];
  *(uint2*)&dst[(size_t)(n0 + r) * 1024 + k0 + c4] = io.u;
}

// ---------------------------------------------------------------------------
__global__ __launch_bounds__(256) void embed_kernel(
    const int* __restrict__ ids, const float* __restrict__ tok,
    const float* __restrict__ pos, float* __restrict__ x) {
  const int t = blockIdx.x;
  const int id = ids[t];
  const int p = t & 1023;
#pragma unroll
  for (int i = threadIdx.x; i < 768; i += 256)
    x[(size_t)t * 768 + i] = tok[(size_t)id * 768 + i] + pos[(size_t)p * 768 + i];
}

// LayerNorm (two-pass, eps 1e-5), fp32 in -> bf16 out. One block per row.
__global__ __launch_bounds__(256) void layernorm_bf16(
    const float* __restrict__ x, const float* __restrict__ wgt,
    const float* __restrict__ bias, unsigned short* __restrict__ out) {
  const int row = blockIdx.x, tid = threadIdx.x;
  const float* xr = x + (size_t)row * 768;
  __shared__ float red[256];
  float v[3], s = 0.f;
#pragma unroll
  for (int i = 0; i < 3; ++i) { v[i] = xr[tid + i * 256]; s += v[i]; }
  red[tid] = s; __syncthreads();
  for (int o = 128; o > 0; o >>= 1) { if (tid < o) red[tid] += red[tid + o]; __syncthreads(); }
  const float mean = red[0] * (1.0f / 768.0f);
  __syncthreads();
  float s2 = 0.f;
#pragma unroll
  for (int i = 0; i < 3; ++i) { const float d = v[i] - mean; s2 += d * d; }
  red[tid] = s2; __syncthreads();
  for (int o = 128; o > 0; o >>= 1) { if (tid < o) red[tid] += red[tid + o]; __syncthreads(); }
  const float rstd = rsqrtf(red[0] * (1.0f / 768.0f) + 1e-5f);
#pragma unroll
  for (int i = 0; i < 3; ++i) {
    const int c = tid + i * 256;
    out[(size_t)row * 768 + c] = f2bf((v[i] - mean) * rstd * wgt[c] + bias[c]);
  }
}

// Row softmax: fp32 scores -> bf16 probabilities. One block per row.
__global__ __launch_bounds__(256) void softmax_bf16(
    const float* __restrict__ sc, unsigned short* __restrict__ att, int n) {
  const size_t row = blockIdx.x;
  const float* p = sc + row * (size_t)n;
  unsigned short* o = att + row * (size_t)n;
  __shared__ float red[256];
  const int tid = threadIdx.x;
  float m = -3.0e38f;
  for (int i = tid; i < n; i += 256) m = fmaxf(m, p[i]);
  red[tid] = m; __syncthreads();
  for (int s = 128; s > 0; s >>= 1) { if (tid < s) red[tid] = fmaxf(red[tid], red[tid + s]); __syncthreads(); }
  m = red[0]; __syncthreads();
  float sum = 0.f;
  for (int i = tid; i < n; i += 256) sum += expf(p[i] - m);
  red[tid] = sum; __syncthreads();
  for (int s = 128; s > 0; s >>= 1) { if (tid < s) red[tid] += red[tid + s]; __syncthreads(); }
  const float inv = 1.0f / red[0];
  for (int i = tid; i < n; i += 256) o[i] = f2bf(expf(p[i] - m) * inv);
}

// ---------------------------------------------------------------------------
extern "C" void kernel_launch(void* const* d_in, const int* in_sizes, int n_in,
                              void* d_out, int out_size, void* d_ws, size_t ws_size,
                              hipStream_t stream) {
  constexpr int kB = 2, kS = 1024, kT = kB * kS, kD = 768, kH = 12, kDH = 64;
  constexpr int kFF = 3072, kV = 50257, kL = 12;
  typedef long long ll;

  const bool idsFirst = (in_sizes[0] == kT);
  const int pb = idsFirst ? 1 : 0;
  const int*   ids     = (const int*)d_in[idsFirst ? 0 : (n_in - 1)];
  const float* tok_emb = (const float*)d_in[pb + 0];
  const float* pos_emb = (const float*)d_in[pb + 1];
  const float* ln1_w   = (const float*)d_in[pb + 2];
  const float* ln1_b   = (const float*)d_in[pb + 3];
  const float* Wa      = (const float*)d_in[pb + 4];
  const float* ba      = (const float*)d_in[pb + 5];
  const float* Wo      = (const float*)d_in[pb + 6];
  const float* bo      = (const float*)d_in[pb + 7];
  const float* ln2_w   = (const float*)d_in[pb + 8];
  const float* ln2_b   = (const float*)d_in[pb + 9];
  const float* W1      = (const float*)d_in[pb + 10];
  const float* b1      = (const float*)d_in[pb + 11];
  const float* W2      = (const float*)d_in[pb + 12];
  const float* b2      = (const float*)d_in[pb + 13];
  const float* lnf_w   = (const float*)d_in[pb + 14];
  const float* lnf_b   = (const float*)d_in[pb + 15];
  const float* lm_head = (const float*)d_in[pb + 16];
  (void)out_size;

  // --- workspace carve-out -------------------------------------------------
  char* wp = (char*)d_ws;
  auto carve = [&](size_t bytes) {
    char* p = wp; wp += (bytes + 255) & ~(size_t)255; return p;
  };
  float*          x      = (float*)carve((size_t)kT * kD * 4);
  unsigned short* hbf    = (unsigned short*)carve((size_t)kT * kD * 2);
  unsigned short* qkvb   = (unsigned short*)carve((size_t)kT * 3 * kD * 2);
  unsigned short* mlpb   = (unsigned short*)carve((size_t)kT * kFF * 2);
  unsigned short* vT     = (unsigned short*)carve((size_t)kB * kH * kDH * kS * 2);
  float*          scores = (float*)carve((size_t)kB * kH * kS * kS * 4);
  unsigned short* att    = (unsigned short*)carve((size_t)kB * kH * kS * kS * 2);

  const size_t wWa = (size_t)kL * kD * 3 * kD, wWo = (size_t)kL * kD * kD;
  const size_t wW1 = (size_t)kL * kD * kFF,    wW2 = (size_t)kL * kFF * kD;
  const size_t wLm = (size_t)kV * kD;
  const size_t usedBase = (size_t)(wp - (char*)d_ws);
  const size_t needW = ((wWa + wWo + wW1 + wW2 + wLm) * 2) + 5 * 256;
  const bool preconv = (ws_size >= usedBase + needW);

  unsigned short *WaT = nullptr, *WoT = nullptr, *W1T = nullptr,
                 *W2T = nullptr, *LmB = nullptr;
  if (preconv) {
    WaT = (unsigned short*)carve(wWa * 2);
    WoT = (unsigned short*)carve(wWo * 2);
    W1T = (unsigned short*)carve(wW1 * 2);
    W2T = (unsigned short*)carve(wW2 * 2);
    LmB = (unsigned short*)carve(wLm * 2);
    // One-time fp32 -> bf16 [N,K] weight conversion (~0.75 GB HBM, ~32 us)
    transpose_cvt_f32<<<dim3(3 * kD / 32, kD / 32, kL), 256, 0, stream>>>(
        Wa, WaT, kD, 3 * kD, (ll)kD * 3 * kD, (ll)kD * 3 * kD);
    transpose_cvt_f32<<<dim3(kD / 32, kD / 32, kL), 256, 0, stream>>>(
        Wo, WoT, kD, kD, (ll)kD * kD, (ll)kD * kD);
    transpose_cvt_f32<<<dim3(kFF / 32, kD / 32, kL), 256, 0, stream>>>(
        W1, W1T, kD, kFF, (ll)kD * kFF, (ll)kD * kFF);
    transpose_cvt_f32<<<dim3(kD / 32, kFF / 32, kL), 256, 0, stream>>>(
        W2, W2T, kFF, kD, (ll)kFF * kD, (ll)kFF * kD);
    copy_cvt_f32<<<(int)((wLm / 4 + 255) / 256), 256, 0, stream>>>(
        lm_head, LmB, (int)(wLm / 4));
  }

  embed_kernel<<<kT, 256, 0, stream>>>(ids, tok_emb, pos_emb, x);

  const dim3 blk(256);
  for (int l = 0; l < kL; ++l) {
    layernorm_bf16<<<kT, blk, 0, stream>>>(x, ln1_w + (size_t)l * kD,
                                           ln1_b + (size_t)l * kD, hbf);
    // qkv = h @ Wa + ba (bf16 out)
    if (preconv)
      gemm_wmma<unsigned short, F_BTRANS | F_BIAS | F_BF16>
          <<<dim3(3 * kD / 128, kT / 256, 1), blk, 0, stream>>>(
              hbf, WaT + (size_t)l * kD * 3 * kD, ba + (size_t)l * 3 * kD,
              nullptr, qkvb, nullptr, 3 * kD, kD, kD, kD, 3 * kD,
              0, 0, 0, 0, 0, 0, 1, 1.0f);
    else
      gemm_wmma<float, F_BIAS | F_BF16>
          <<<dim3(3 * kD / 128, kT / 256, 1), blk, 0, stream>>>(
              hbf, Wa + (size_t)l * kD * 3 * kD, ba + (size_t)l * 3 * kD,
              nullptr, qkvb, nullptr, 3 * kD, kD, kD, 3 * kD, 3 * kD,
              0, 0, 0, 0, 0, 0, 1, 1.0f);
    // vT[b,h][dh][key] for the AV GEMM's contiguous-K path
    transpose_v<<<dim3(kDH / 32, kS / 32, kB * kH), blk, 0, stream>>>(qkvb, vT);
    // scores = (q @ k^T)/8, causal (batched over B*H). k is [key][dim]=[N][K].
    gemm_wmma<unsigned short, F_BTRANS | F_CAUSAL>
        <<<dim3(kS / 128, kS / 256, kB * kH), blk, 0, stream>>>(
            qkvb, qkvb + kD, nullptr, scores, nullptr, nullptr,
            kS, kDH, 3 * kD, 3 * kD, kS,
            (ll)kS * 3 * kD, 64, (ll)kS * 3 * kD, 64,
            (ll)kH * kS * kS, (ll)kS * kS, kH, 0.125f);
    softmax_bf16<<<kB * kH * kS, blk, 0, stream>>>(scores, att, kS);
    // o = att @ v : B = vT [dh][key] = [N][K]; N=64 needs guard
    gemm_wmma<unsigned short, F_BTRANS | F_BF16 | F_NGUARD>
        <<<dim3(1, kS / 256, kB * kH), blk, 0, stream>>>(
            att, vT, nullptr, nullptr, hbf, nullptr,
            kDH, kS, kS, kS, kD,
            (ll)kH * kS * kS, (ll)kS * kS, (ll)kH * kDH * kS, (ll)kDH * kS,
            (ll)kS * kD, 64, kH, 1.0f);
    // x += o @ Wo + bo
    if (preconv)
      gemm_wmma<unsigned short, F_BTRANS | F_BIAS | F_RES>
          <<<dim3(kD / 128, kT / 256, 1), blk, 0, stream>>>(
              hbf, WoT + (size_t)l * kD * kD, bo + (size_t)l * kD,
              x, nullptr, x, kD, kD, kD, kD, kD, 0, 0, 0, 0, 0, 0, 1, 1.0f);
    else
      gemm_wmma<float, F_BIAS | F_RES>
          <<<dim3(kD / 128, kT / 256, 1), blk, 0, stream>>>(
              hbf, Wo + (size_t)l * kD * kD, bo + (size_t)l * kD,
              x, nullptr, x, kD, kD, kD, kD, kD, 0, 0, 0, 0, 0, 0, 1, 1.0f);
    layernorm_bf16<<<kT, blk, 0, stream>>>(x, ln2_w + (size_t)l * kD,
                                           ln2_b + (size_t)l * kD, hbf);
    // m = gelu(h2 @ W1 + b1)
    if (preconv)
      gemm_wmma<unsigned short, F_BTRANS | F_BIAS | F_GELU | F_BF16>
          <<<dim3(kFF / 128, kT / 256, 1), blk, 0, stream>>>(
              hbf, W1T + (size_t)l * kD * kFF, b1 + (size_t)l * kFF,
              nullptr, mlpb, nullptr, kFF, kD, kD, kD, kFF,
              0, 0, 0, 0, 0, 0, 1, 1.0f);
    else
      gemm_wmma<float, F_BIAS | F_GELU | F_BF16>
          <<<dim3(kFF / 128, kT / 256, 1), blk, 0, stream>>>(
              hbf, W1 + (size_t)l * kD * kFF, b1 + (size_t)l * kFF,
              nullptr, mlpb, nullptr, kFF, kD, kD, kFF, kFF,
              0, 0, 0, 0, 0, 0, 1, 1.0f);
    // x += m @ W2 + b2
    if (preconv)
      gemm_wmma<unsigned short, F_BTRANS | F_BIAS | F_RES>
          <<<dim3(kD / 128, kT / 256, 1), blk, 0, stream>>>(
              mlpb, W2T + (size_t)l * kFF * kD, b2 + (size_t)l * kD,
              x, nullptr, x, kD, kFF, kFF, kFF, kD, 0, 0, 0, 0, 0, 0, 1, 1.0f);
    else
      gemm_wmma<float, F_BIAS | F_RES>
          <<<dim3(kD / 128, kT / 256, 1), blk, 0, stream>>>(
              mlpb, W2 + (size_t)l * kFF * kD, b2 + (size_t)l * kD,
              x, nullptr, x, kD, kFF, kFF, kD, kD, 0, 0, 0, 0, 0, 0, 1, 1.0f);
  }

  layernorm_bf16<<<kT, blk, 0, stream>>>(x, lnf_w, lnf_b, hbf);
  // logits = h @ lm_head^T : lm_head is [V,D] = [N,K] already
  if (preconv)
    gemm_wmma<unsigned short, F_BTRANS | F_NGUARD>
        <<<dim3((kV + 127) / 128, kT / 256, 1), blk, 0, stream>>>(
            hbf, LmB, nullptr, (float*)d_out, nullptr, nullptr,
            kV, kD, kD, kD, kV, 0, 0, 0, 0, 0, 0, 1, 1.0f);
  else
    gemm_wmma<float, F_BTRANS | F_NGUARD>
        <<<dim3((kV + 127) / 128, kT / 256, 1), blk, 0, stream>>>(
            hbf, lm_head, nullptr, (float*)d_out, nullptr, nullptr,
            kV, kD, kD, kD, kV, 0, 0, 0, 0, 0, 0, 1, 1.0f);
}
